// LSTMSoftAttentionCore_34651796144398
// MI455X (gfx1250) — matrix operand order repaired and checked by reading, not compiled, not run on hardware
//
#include <hip/hip_runtime.h>
#include <math.h>

typedef __attribute__((ext_vector_type(16))) __bf16 v16bf;
typedef __attribute__((ext_vector_type(8)))  float  v8f;

union Frag16 {
    v16bf v;
    uint4 q[2];
};

// ---------------------------------------------------------------- constants
#define BSZ   256
#define LSEQ  196
#define CDIM  2048
#define EDIM  512
#define HDIM  512
#define ADIM  512
#define ROWS  (BSZ * LSEQ)          // 50176 rows of the attention GEMM
#define KCAT  (EDIM + HDIM + CDIM)  // 3072, concatenated K for the gate GEMM
#define N4H   (4 * HDIM)            // 2048

// workspace offsets (bytes, all 16B-aligned)
static const size_t OFF_WABF   = 0;           // Wa  bf16: 512*2048*2   = 2,097,152
static const size_t OFF_HLIN   = 2097152;     // h_linear fp32 256*512  =   524,288
static const size_t OFF_SCORES = 2621440;     // scores fp32 50176      =   200,704
static const size_t OFF_CONV   = 2822144;     // softmax weights 50176  =   200,704
static const size_t OFF_Z      = 3022848;     // z fp32 256*2048        = 2,097,152
static const size_t OFF_XP     = 5120000;     // X packed bf16 256*3072 = 1,572,864
static const size_t OFF_WP     = 6692864;     // W packed bf16 2048*3072= 12,582,912
static const size_t OFF_S      = 19275776;    // gate sums fp32 256*2048= 2,097,152

// ---------------------------------------------------------------- helpers
__device__ __forceinline__ unsigned short f2bf(float f) {
    unsigned int u = __float_as_uint(f);
    unsigned int r = u + 0x7FFFu + ((u >> 16) & 1u);   // round-to-nearest-even
    return (unsigned short)(r >> 16);
}
__device__ __forceinline__ unsigned int pack2bf(float a, float b) {
    return (unsigned int)f2bf(a) | ((unsigned int)f2bf(b) << 16);
}
__device__ __forceinline__ float sigmoidf(float x) {
    return 1.0f / (1.0f + expf(-x));
}

// ---------------------------------------------------------------- prep kernels
__global__ void pack_wa_bf16(const float* __restrict__ src,
                             unsigned short* __restrict__ dst, int n) {
    int i = blockIdx.x * blockDim.x + threadIdx.x;
    if (i < n) dst[i] = f2bf(src[i]);
}

// h_linear = pre_h @ Wha^T + bha   -> (256, 512); tiny GEMM, VALU is fine
__global__ void hlin_kernel(const float* __restrict__ pre_h,
                            const float* __restrict__ Wha,
                            const float* __restrict__ bha,
                            float* __restrict__ hlin) {
    int idx = blockIdx.x * blockDim.x + threadIdx.x;   // 256*512
    int b = idx >> 9;
    int a = idx & 511;
    const float* hp = pre_h + b * HDIM;
    const float* wp = Wha + (size_t)a * HDIM;
    float acc = 0.f;
    #pragma unroll 4
    for (int k = 0; k < HDIM; ++k) acc += hp[k] * wp[k];
    hlin[idx] = acc + bha[a];
}

// pack X = [xt | pre_h | z] -> bf16 (256, 3072)
__global__ void pack_xp(const float* __restrict__ xt,
                        const float* __restrict__ pre_h,
                        const float* __restrict__ z,
                        unsigned short* __restrict__ Xp) {
    int idx = blockIdx.x * blockDim.x + threadIdx.x;   // 256*3072
    int b = idx / KCAT;
    int c = idx - b * KCAT;
    float v;
    if (c < EDIM)              v = xt[b * EDIM + c];
    else if (c < EDIM + HDIM)  v = pre_h[b * HDIM + (c - EDIM)];
    else                       v = z[(size_t)b * CDIM + (c - EDIM - HDIM)];
    Xp[idx] = f2bf(v);
}

// pack W = [Wi | Wh | Wz] -> bf16 (2048, 3072)
__global__ void pack_wp(const float* __restrict__ Wi,
                        const float* __restrict__ Wh,
                        const float* __restrict__ Wz,
                        unsigned short* __restrict__ Wp) {
    int idx = blockIdx.x * blockDim.x + threadIdx.x;   // 2048*3072
    int j = idx / KCAT;
    int c = idx - j * KCAT;
    float v;
    if (c < EDIM)              v = Wi[(size_t)j * EDIM + c];
    else if (c < EDIM + HDIM)  v = Wh[(size_t)j * HDIM + (c - EDIM)];
    else                       v = Wz[(size_t)j * CDIM + (c - EDIM - HDIM)];
    Wp[idx] = f2bf(v);
}

// ---------------------------------------------------------------- kernel B:
// fused  scores[r] = sum_a tanh( att[r,:]@Wa[a,:] + ba[a] + hlin[b(r),a] ) * Wo[a] + bo
// One block = 32 GEMM rows (2 M-tiles), 8 waves, each wave owns 64 of the 512
// A-columns. Each B fragment (L2-resident) feeds two WMMAs.
__global__ __launch_bounds__(256) void attn_scores_kernel(
        const float* __restrict__ att,            // (50176, 2048)
        const unsigned short* __restrict__ WaBf,  // (512, 2048) bf16
        const float* __restrict__ hlin,           // (256, 512)
        const float* __restrict__ ba,             // (512)
        const float* __restrict__ Wo,             // (512)
        const float* __restrict__ bo,             // (1)
        float* __restrict__ scores)               // (50176)
{
    __shared__ unsigned short As[32 * 128];       // staged A tile, bf16 (8 KB)
    __shared__ float partial[32];

    const int tid    = threadIdx.x;
    const int lane   = tid & 31;
    const int wave   = tid >> 5;                  // 0..7
    const int m      = lane & 15;
    const int hiHalf = lane >> 4;                 // 0|1
    const int r0     = blockIdx.x * 32;
    const int nwave  = wave * 64;

    if (tid < 32) partial[tid] = 0.0f;

    v8f acc0[4], acc1[4];
    #pragma unroll
    for (int f = 0; f < 4; ++f) { acc0[f] = v8f{}; acc1[f] = v8f{}; }

    for (int kk = 0; kk < CDIM; kk += 128) {
        __syncthreads();
        {   // stage 32x128 fp32 -> bf16 into LDS; thread converts 16 floats
            int row = tid >> 3;            // 0..31
            int col = (tid & 7) * 16;      // 0..112
            const float* src = att + (size_t)(r0 + row) * CDIM + kk + col;
            float4 f0 = ((const float4*)src)[0];
            float4 f1 = ((const float4*)src)[1];
            float4 f2 = ((const float4*)src)[2];
            float4 f3 = ((const float4*)src)[3];
            uint4 p0, p1;
            p0.x = pack2bf(f0.x, f0.y);  p0.y = pack2bf(f0.z, f0.w);
            p0.z = pack2bf(f1.x, f1.y);  p0.w = pack2bf(f1.z, f1.w);
            p1.x = pack2bf(f2.x, f2.y);  p1.y = pack2bf(f2.z, f2.w);
            p1.z = pack2bf(f3.x, f3.y);  p1.w = pack2bf(f3.z, f3.w);
            *((uint4*)&As[row * 128 + col])     = p0;
            *((uint4*)&As[row * 128 + col + 8]) = p1;
        }
        __syncthreads();

        #pragma unroll
        for (int k2 = 0; k2 < 128; k2 += 32) {
            // A fragments for both M-tiles:
            // lanes 0-15 -> K {0..7,16..23}; lanes 16-31 -> K {8..15,24..31}
            Frag16 a0, a1;
            const unsigned short* ap0 = &As[m * 128 + k2 + hiHalf * 8];
            const unsigned short* ap1 = &As[(m + 16) * 128 + k2 + hiHalf * 8];
            a0.q[0] = *(const uint4*)ap0;
            a0.q[1] = *(const uint4*)(ap0 + 16);
            a1.q[0] = *(const uint4*)ap1;
            a1.q[1] = *(const uint4*)(ap1 + 16);
            #pragma unroll
            for (int f = 0; f < 4; ++f) {
                int n = nwave + f * 16 + m;
                // B fragment: lane n, K {0..7,16..23} (lo) / {8..15,24..31} (hi)
                const unsigned short* bp =
                    WaBf + (size_t)n * CDIM + kk + k2 + hiHalf * 8;
                Frag16 bfr;
                bfr.q[0] = *(const uint4*)bp;
                bfr.q[1] = *(const uint4*)(bp + 16);
                acc0[f] = __builtin_amdgcn_wmma_f32_16x16x32_bf16(
                    false, a0.v, false, bfr.v, (short)0, acc0[f], false, false);
                acc1[f] = __builtin_amdgcn_wmma_f32_16x16x32_bf16(
                    false, a1.v, false, bfr.v, (short)0, acc1[f], false, false);
            }
        }
    }

    __syncthreads();
    // epilogue: tanh(acc + ba + hlin) * Wo, reduce over all 512 columns
    float s0[8], s1[8];
    #pragma unroll
    for (int i = 0; i < 8; ++i) { s0[i] = 0.f; s1[i] = 0.f; }
    #pragma unroll
    for (int f = 0; f < 4; ++f) {
        int n = nwave + f * 16 + m;
        float ban = ba[n];
        float won = Wo[n];
        #pragma unroll
        for (int i = 0; i < 8; ++i) {
            int mrow = i + hiHalf * 8;
            {
                int r = r0 + mrow;
                int b = r / LSEQ;
                float v = acc0[f][i] + ban + hlin[b * ADIM + n];
                s0[i] += tanhf(v) * won;
            }
            {
                int r = r0 + 16 + mrow;
                int b = r / LSEQ;
                float v = acc1[f][i] + ban + hlin[b * ADIM + n];
                s1[i] += tanhf(v) * won;
            }
        }
    }
    #pragma unroll
    for (int i = 0; i < 8; ++i) {
        atomicAdd(&partial[i + hiHalf * 8], s0[i]);        // ds_add_f32
        atomicAdd(&partial[16 + i + hiHalf * 8], s1[i]);
    }
    __syncthreads();
    if (tid < 32) scores[r0 + tid] = partial[tid] + bo[0];
}

// ---------------------------------------------------------------- softmax over L
__global__ __launch_bounds__(256) void softmax_kernel(
        const float* __restrict__ scores, float* __restrict__ w) {
    __shared__ float red[256];
    int b = blockIdx.x;
    int t = threadIdx.x;
    float v = (t < LSEQ) ? scores[b * LSEQ + t] : -INFINITY;
    red[t] = v;
    __syncthreads();
    for (int s = 128; s > 0; s >>= 1) {
        if (t < s) red[t] = fmaxf(red[t], red[t + s]);
        __syncthreads();
    }
    float mx = red[0];
    __syncthreads();
    float e = (t < LSEQ) ? expf(v - mx) : 0.f;
    red[t] = e;
    __syncthreads();
    for (int s = 128; s > 0; s >>= 1) {
        if (t < s) red[t] += red[t + s];
        __syncthreads();
    }
    float sum = red[0];
    if (t < LSEQ) w[b * LSEQ + t] = e / sum;
}

// ---------------------------------------------------------------- z[b,c] = sum_l att[b,l,c]*w[b,l]
// pure bandwidth pass; float4 per thread, w staged in LDS
__global__ __launch_bounds__(256) void zred_kernel(
        const float* __restrict__ att, const float* __restrict__ w,
        float* __restrict__ z) {
    __shared__ float ws[LSEQ];
    int b = blockIdx.x;
    int c4 = blockIdx.y * 256 + threadIdx.x;          // float4 index 0..511
    if (threadIdx.x < LSEQ) ws[threadIdx.x] = w[b * LSEQ + threadIdx.x];
    __syncthreads();
    const float4* base = (const float4*)(att + (size_t)b * LSEQ * CDIM) + c4;
    float4 acc = make_float4(0.f, 0.f, 0.f, 0.f);
    for (int l = 0; l < LSEQ; ++l) {
        float4 v = base[(size_t)l * (CDIM / 4)];
        float wl = ws[l];
        acc.x += v.x * wl; acc.y += v.y * wl;
        acc.z += v.z * wl; acc.w += v.w * wl;
    }
    ((float4*)(z + (size_t)b * CDIM))[c4] = acc;
}

// ---------------------------------------------------------------- kernel E:
// S = Xp @ Wp^T   (256x3072 bf16) x (2048x3072 bf16) -> fp32 (256, 2048)
// Same 32-row M-tile scheme: each B fragment feeds two WMMAs.
__global__ __launch_bounds__(256) void gate_gemm_kernel(
        const unsigned short* __restrict__ Xp,
        const unsigned short* __restrict__ Wp,
        float* __restrict__ out) {
    __shared__ unsigned short As[32 * 128];

    const int tid    = threadIdx.x;
    const int lane   = tid & 31;
    const int wave   = tid >> 5;
    const int m      = lane & 15;
    const int hiHalf = lane >> 4;
    const int r0     = blockIdx.x * 32;               // 8 M-tiles of 32
    const int n0     = blockIdx.y * 512 + wave * 64;  // 4 N-blocks of 512

    v8f acc0[4], acc1[4];
    #pragma unroll
    for (int f = 0; f < 4; ++f) { acc0[f] = v8f{}; acc1[f] = v8f{}; }

    for (int kk = 0; kk < KCAT; kk += 128) {
        __syncthreads();
        {   // copy bf16 X tile into LDS (already bf16): 16 ushorts / thread
            int row = tid >> 3;
            int col = (tid & 7) * 16;
            const unsigned short* src = Xp + (size_t)(r0 + row) * KCAT + kk + col;
            *((uint4*)&As[row * 128 + col])     = ((const uint4*)src)[0];
            *((uint4*)&As[row * 128 + col + 8]) = ((const uint4*)src)[1];
        }
        __syncthreads();

        #pragma unroll
        for (int k2 = 0; k2 < 128; k2 += 32) {
            Frag16 a0, a1;
            const unsigned short* ap0 = &As[m * 128 + k2 + hiHalf * 8];
            const unsigned short* ap1 = &As[(m + 16) * 128 + k2 + hiHalf * 8];
            a0.q[0] = *(const uint4*)ap0;
            a0.q[1] = *(const uint4*)(ap0 + 16);
            a1.q[0] = *(const uint4*)ap1;
            a1.q[1] = *(const uint4*)(ap1 + 16);
            #pragma unroll
            for (int f = 0; f < 4; ++f) {
                int n = n0 + f * 16 + m;
                const unsigned short* bp =
                    Wp + (size_t)n * KCAT + kk + k2 + hiHalf * 8;
                Frag16 bfr;
                bfr.q[0] = *(const uint4*)bp;
                bfr.q[1] = *(const uint4*)(bp + 16);
                acc0[f] = __builtin_amdgcn_wmma_f32_16x16x32_bf16(
                    false, a0.v, false, bfr.v, (short)0, acc0[f], false, false);
                acc1[f] = __builtin_amdgcn_wmma_f32_16x16x32_bf16(
                    false, a1.v, false, bfr.v, (short)0, acc1[f], false, false);
            }
        }
    }

    #pragma unroll
    for (int f = 0; f < 4; ++f) {
        int n = n0 + f * 16 + m;
        #pragma unroll
        for (int i = 0; i < 8; ++i) {
            int mrow = i + hiHalf * 8;
            out[(size_t)(r0 + mrow) * N4H + n]      = acc0[f][i];
            out[(size_t)(r0 + 16 + mrow) * N4H + n] = acc1[f][i];
        }
    }
}

// ---------------------------------------------------------------- final LSTM elementwise
__global__ __launch_bounds__(256) void lstm_kernel(
        const float* __restrict__ S,
        const float* __restrict__ bi, const float* __restrict__ bh,
        const float* __restrict__ bz, const float* __restrict__ c0,
        float* __restrict__ out) {
    int idx = blockIdx.x * blockDim.x + threadIdx.x;  // 256*512
    int b = idx >> 9;
    int h = idx & 511;
    const float* Sb = S + (size_t)b * N4H;
    float gi = Sb[h]        + bi[h]        + bh[h]        + bz[h];
    float gf = Sb[h + 512]  + bi[h + 512]  + bh[h + 512]  + bz[h + 512];
    float go = Sb[h + 1024] + bi[h + 1024] + bh[h + 1024] + bz[h + 1024];
    float gt = Sb[h + 1536] + bi[h + 1536] + bh[h + 1536] + bz[h + 1536];
    float ig = sigmoidf(gi);
    float fg = sigmoidf(gf);
    float og = sigmoidf(go);
    float it = tanhf(gt);
    float pc = c0[idx];
    float nc = fg * pc + ig * it;
    float nh = og * tanhf(nc);
    out[idx] = nh;                    // next_h
    out[BSZ * HDIM + idx] = nc;       // next_c
}

// ---------------------------------------------------------------- launch
extern "C" void kernel_launch(void* const* d_in, const int* in_sizes, int n_in,
                              void* d_out, int out_size, void* d_ws, size_t ws_size,
                              hipStream_t stream) {
    const float* xt    = (const float*)d_in[0];
    const float* att   = (const float*)d_in[1];
    const float* h0    = (const float*)d_in[2];
    const float* c0    = (const float*)d_in[3];
    const float* Wi    = (const float*)d_in[4];
    const float* bi    = (const float*)d_in[5];
    const float* Wh    = (const float*)d_in[6];
    const float* bh    = (const float*)d_in[7];
    const float* Wz    = (const float*)d_in[8];
    const float* bz    = (const float*)d_in[9];
    const float* Wa    = (const float*)d_in[10];
    const float* ba    = (const float*)d_in[11];
    const float* Wha   = (const float*)d_in[12];
    const float* bha   = (const float*)d_in[13];
    const float* Wo    = (const float*)d_in[14];
    const float* bo    = (const float*)d_in[15];
    float* out = (float*)d_out;
    char*  ws  = (char*)d_ws;

    unsigned short* WaBf = (unsigned short*)(ws + OFF_WABF);
    float* hlin          = (float*)(ws + OFF_HLIN);
    float* scores        = (float*)(ws + OFF_SCORES);
    float* conv          = (float*)(ws + OFF_CONV);
    float* z             = (float*)(ws + OFF_Z);
    unsigned short* Xp   = (unsigned short*)(ws + OFF_XP);
    unsigned short* Wp   = (unsigned short*)(ws + OFF_WP);
    float* S             = (float*)(ws + OFF_S);

    // prep (independent of scores path, run up front)
    pack_wa_bf16<<<(ADIM * CDIM) / 256, 256, 0, stream>>>(Wa, WaBf, ADIM * CDIM);
    hlin_kernel<<<(BSZ * ADIM) / 256, 256, 0, stream>>>(h0, Wha, bha, hlin);
    pack_wp<<<(N4H * KCAT) / 256, 256, 0, stream>>>(Wi, Wh, Wz, Wp);

    // fused attention-score GEMM (dominant op, WMMA bf16)
    attn_scores_kernel<<<ROWS / 32, 256, 0, stream>>>(att, WaBf, hlin, ba, Wo, bo, scores);

    // softmax + weighted pooling of att_seq
    softmax_kernel<<<BSZ, 256, 0, stream>>>(scores, conv);
    zred_kernel<<<dim3(BSZ, 2), 256, 0, stream>>>(att, conv, z);

    // gate GEMM (WMMA bf16) + LSTM cell
    pack_xp<<<(BSZ * KCAT) / 256, 256, 0, stream>>>(xt, h0, z, Xp);
    gate_gemm_kernel<<<dim3(8, 4), 256, 0, stream>>>(Xp, Wp, S);
    lstm_kernel<<<(BSZ * HDIM) / 256, 256, 0, stream>>>(S, bi, bh, bz, c0, out);
}
